// TriangleAttention_515396076358
// MI455X (gfx1250) — matrix operand order repaired and checked by reading, not compiled, not run on hardware
//
#include <hip/hip_runtime.h>
#include <hip/hip_bf16.h>

typedef __attribute__((ext_vector_type(16))) _Float16 v16h;
typedef __attribute__((ext_vector_type(8)))  float    v8f;
typedef __attribute__((ext_vector_type(4)))  _Float16 v4h;

#define TA_S 256
#define TA_D 128
#define TA_H 4
#define TA_DH 32
static constexpr float RSCALE = 0.17677669529663687f; // 1/sqrt(32)

// ---------------------------------------------------------------------------
// CDNA5 async global->LDS copy (ASYNCcnt path), 16B per active lane.
// Inline asm per CDNA5 ISA (portable across toolchains).
// ---------------------------------------------------------------------------
__device__ inline void async_ld_b128(void* lds_dst, const void* g_src) {
  // Shared-aperture flat addresses carry the LDS byte offset in the low 32
  // bits (isLDS check covers [63:32]); VDST operand takes that 32-bit addr.
  unsigned l = (unsigned)(uintptr_t)lds_dst;
  asm volatile("global_load_async_to_lds_b128 %0, %1, off"
               :
               : "v"(l), "v"(g_src)
               : "memory");
}
__device__ inline void async_wait0() {
#if __has_builtin(__builtin_amdgcn_s_wait_asynccnt)
  __builtin_amdgcn_s_wait_asynccnt(0);
#else
  asm volatile("s_wait_asynccnt 0x0" ::: "memory");
#endif
}

// ---------------------------------------------------------------------------
// WMMA fragment helpers (wave32; A and B both use contiguous k = 16*(L>>4) + element)
// ---------------------------------------------------------------------------
__device__ inline v8f wmma_f16(v16h a, v16h b, v8f c) {
  // 8 args: (neg_a, A, neg_b, B, c_mod, C, reuse_a, reuse_b)
  return __builtin_amdgcn_wmma_f32_16x16x32_f16(false, a, false, b, (short)0, c,
                                                false, false);
}

// A matrix 16x32 f16: lane L holds row M = L&15; group g = L>>4.
// VGPR v in 0..3 -> K = 2v+p+16g ; VGPR v in 4..7 -> K = 8+2(v-4)+p+16g
__device__ inline v16h load_a_frag(const _Float16* base, int ld, int row0,
                                   int k0, int lane) {
  v16h a;
  const _Float16* p = base + (size_t)(row0 + (lane & 15)) * ld + k0;
  int g8 = (lane >> 4) << 4;
#pragma unroll
  for (int v = 0; v < 4; ++v) {
    a[2 * v]     = p[2 * v + g8];
    a[2 * v + 1] = p[2 * v + g8 + 1];
    a[2 * v + 8] = p[8 + 2 * v + g8];
    a[2 * v + 9] = p[8 + 2 * v + g8 + 1];
  }
  return a;
}

// B matrix 32x16 f16: lane L holds col N = L&15; K = 2v+p+16*(L>>4).
// Source stored N-major: base[n*ld + k]  (i.e. B = W^T with W row-major [N,K])
__device__ inline v16h load_b_frag_nmajor(const _Float16* base, int ld, int n0,
                                          int k0, int lane) {
  v16h b;
  const _Float16* p =
      base + (size_t)(n0 + (lane & 15)) * ld + k0 + ((lane >> 4) << 4);
#pragma unroll
  for (int v = 0; v < 8; ++v) {
    b[2 * v]     = p[2 * v];
    b[2 * v + 1] = p[2 * v + 1];
  }
  return b;
}

// B matrix 32x16 f16 from K-major storage: base[k*ld + n]
__device__ inline v16h load_b_frag_kmajor(const _Float16* base, int ld, int k0,
                                          int n0, int lane) {
  v16h b;
  const _Float16* p =
      base + (size_t)(k0 + ((lane >> 4) << 4)) * ld + n0 + (lane & 15);
#pragma unroll
  for (int v = 0; v < 8; ++v) {
    b[2 * v]     = p[(2 * v) * ld];
    b[2 * v + 1] = p[(2 * v + 1) * ld];
  }
  return b;
}

// ---------------------------------------------------------------------------
// Kernel 1: LayerNorm over D=128, one wave32 per row, output f16
// ---------------------------------------------------------------------------
__global__ __launch_bounds__(256) void ln_kernel(const float* __restrict__ x,
                                                 const float* __restrict__ wln,
                                                 const float* __restrict__ bln,
                                                 _Float16* __restrict__ Xn) {
  const int wid = threadIdx.x >> 5, lane = threadIdx.x & 31;
  const size_t row = (size_t)blockIdx.x * 8 + wid;
  const float4 xv = *(const float4*)(x + row * TA_D + lane * 4);
  float s = xv.x + xv.y + xv.z + xv.w;
#pragma unroll
  for (int m = 16; m >= 1; m >>= 1) s += __shfl_xor(s, m, 32);
  const float mu = s * (1.0f / 128.0f);
  const float d0 = xv.x - mu, d1 = xv.y - mu, d2 = xv.z - mu, d3 = xv.w - mu;
  float ss = d0 * d0 + d1 * d1 + d2 * d2 + d3 * d3;
#pragma unroll
  for (int m = 16; m >= 1; m >>= 1) ss += __shfl_xor(ss, m, 32);
  const float rstd = rsqrtf(ss * (1.0f / 128.0f) + 1e-5f);
  const float4 wv = *(const float4*)(wln + lane * 4);
  const float4 bv = *(const float4*)(bln + lane * 4);
  v4h o;
  o[0] = (_Float16)(d0 * rstd * wv.x + bv.x);
  o[1] = (_Float16)(d1 * rstd * wv.y + bv.y);
  o[2] = (_Float16)(d2 * rstd * wv.z + bv.z);
  o[3] = (_Float16)(d3 * rstd * wv.w + bv.w);
  *(v4h*)(Xn + row * TA_D + lane * 4) = o;
}

// ---------------------------------------------------------------------------
// Kernel 2: weight f32 -> f16 conversion (one shot)
// ---------------------------------------------------------------------------
__global__ void cvt_kernel(const float* wq, const float* wk, const float* wv,
                           const float* wg, const float* wo, const float* wb,
                           _Float16* q16, _Float16* k16, _Float16* v16,
                           _Float16* g16, _Float16* o16, _Float16* b16) {
  const int i = blockIdx.x * 256 + threadIdx.x;
  if (i < 16384) {
    q16[i] = (_Float16)wq[i];
    k16[i] = (_Float16)wk[i];
    v16[i] = (_Float16)wv[i];
    g16[i] = (_Float16)wg[i];
    o16[i] = (_Float16)wo[i];
  }
  if (i < TA_H * TA_D) b16[i] = (_Float16)wb[i];
}

// ---------------------------------------------------------------------------
// Kernel 3: generic WMMA GEMM  C[M,N] = A[M,128] @ W[N,128]^T
// block tile 128M x 64N, 8 waves (4x2), K=128 staged in LDS via async copies
// ---------------------------------------------------------------------------
template <typename OUT_T>
__global__ __launch_bounds__(256) void gemm128(const _Float16* __restrict__ A,
                                               const _Float16* __restrict__ W,
                                               OUT_T* __restrict__ Cout,
                                               int ldc) {
  __shared__ _Float16 Ash[128][136];  // padded stride: conflict-free frag reads
  __shared__ _Float16 Bsh[64][136];
  const int tid = threadIdx.x;
  const int r0 = blockIdx.x * 128;
  const int c0 = blockIdx.y * 64;
#pragma unroll
  for (int it = 0; it < 8; ++it) {
    const int idx = tid + it * 256;  // 2048 x 16B chunks
    const int row = idx >> 4, ch = idx & 15;
    async_ld_b128(&Ash[row][ch * 8], A + (size_t)(r0 + row) * 128 + ch * 8);
  }
#pragma unroll
  for (int it = 0; it < 4; ++it) {
    const int idx = tid + it * 256;  // 1024 x 16B chunks
    const int row = idx >> 4, ch = idx & 15;
    async_ld_b128(&Bsh[row][ch * 8], W + (size_t)(c0 + row) * 128 + ch * 8);
  }
  async_wait0();
  __syncthreads();
  const int wid = tid >> 5, lane = tid & 31;
  const int mbase = (wid >> 1) * 32, nbase = (wid & 1) * 32;
  v8f acc[2][2] = {};
#pragma unroll
  for (int k0 = 0; k0 < 128; k0 += 32) {
    const v16h a0 = load_a_frag(&Ash[0][0], 136, mbase, k0, lane);
    const v16h a1 = load_a_frag(&Ash[0][0], 136, mbase + 16, k0, lane);
    const v16h b0 = load_b_frag_nmajor(&Bsh[0][0], 136, nbase, k0, lane);
    const v16h b1 = load_b_frag_nmajor(&Bsh[0][0], 136, nbase + 16, k0, lane);
    acc[0][0] = wmma_f16(a0, b0, acc[0][0]);
    acc[0][1] = wmma_f16(a0, b1, acc[0][1]);
    acc[1][0] = wmma_f16(a1, b0, acc[1][0]);
    acc[1][1] = wmma_f16(a1, b1, acc[1][1]);
  }
  const int g = lane >> 4, nlo = lane & 15;
#pragma unroll
  for (int mi = 0; mi < 2; ++mi)
#pragma unroll
    for (int ni = 0; ni < 2; ++ni)
#pragma unroll
      for (int v = 0; v < 8; ++v) {
        const int row = r0 + mbase + mi * 16 + v + 8 * g;
        const int col = c0 + nbase + ni * 16 + nlo;
        Cout[(size_t)row * ldc + col] = (OUT_T)acc[mi][ni][v];
      }
}

// ---------------------------------------------------------------------------
// Kernel 4: triangle bias  tb[h, r] = sum_d Xn[r,d] * wb[h,d]
// ---------------------------------------------------------------------------
__global__ __launch_bounds__(256) void tb_kernel(const _Float16* __restrict__ Xn,
                                                 const _Float16* __restrict__ wb,
                                                 float* __restrict__ TB) {
  const int t = blockIdx.x * 256 + threadIdx.x;  // 4*65536 total
  const int h = t & 3;
  const size_t r = (size_t)(t >> 2);
  const _Float16* xr = Xn + r * TA_D;
  const _Float16* wr = wb + h * TA_D;
  float acc = 0.f;
#pragma unroll 8
  for (int d = 0; d < TA_D; ++d) acc += (float)xr[d] * (float)wr[d];
  TB[(size_t)h * (TA_S * TA_S) + r] = acc;
}

// ---------------------------------------------------------------------------
// Kernel 5: attention per (i, h, 32-row j-tile). 4 waves / 128 threads.
// async-staged K/V/Q -> WMMA scores -> softmax -> WMMA P@V -> sigmoid gate.
// ---------------------------------------------------------------------------
__global__ __launch_bounds__(128) void attn_kernel(
    const _Float16* __restrict__ Qb, const _Float16* __restrict__ Kb,
    const _Float16* __restrict__ Vb, const _Float16* __restrict__ Gb,
    const float* __restrict__ TB, _Float16* __restrict__ OG) {
  __shared__ _Float16 Ksh[TA_S * 40];   // [k][d], padded stride 40
  __shared__ _Float16 Vsh[TA_S * 40];   // [k][d]
  __shared__ _Float16 Qsh[32 * 40];     // [j][d]
  __shared__ _Float16 Psh[32 * 264];    // [j][k], padded stride 264
  __shared__ float tbs[TA_S];
  __shared__ float redm[32][2];
  __shared__ float reds[32][2];

  const int b = blockIdx.x;
  const int i = b >> 5;
  const int h = (b >> 3) & 3;
  const int j0 = (b & 7) * 32;
  const int tid = threadIdx.x;
  const size_t rowbase = (size_t)i * TA_S;

  // stage K, V head slices (256 rows x 32 halves) + Q tile, all async->LDS
#pragma unroll
  for (int it = 0; it < 8; ++it) {
    const int idx = tid + it * 128;  // 1024 x 16B chunks
    const int kk = idx >> 2, c = idx & 3;
    const size_t src = (rowbase + kk) * TA_D + h * TA_DH + c * 8;
    async_ld_b128(&Ksh[kk * 40 + c * 8], Kb + src);
    async_ld_b128(&Vsh[kk * 40 + c * 8], Vb + src);
  }
  {  // Q tile: 32 rows x 32 halves = 128 chunks, one per thread
    const int jj = tid >> 2, c = tid & 3;
    async_ld_b128(&Qsh[jj * 40 + c * 8],
                  Qb + (rowbase + j0 + jj) * TA_D + h * TA_DH + c * 8);
  }
  if (tid < 64)
    *(float4*)&tbs[tid * 4] =
        *(const float4*)(TB + (size_t)h * (TA_S * TA_S) + rowbase + tid * 4);
  async_wait0();
  __syncthreads();

  const int wid = tid >> 5, lane = tid & 31;
  const int mt = wid >> 1;       // which 16-row M tile (j)
  const int nh = wid & 1;        // which 128-col half (k)
  const int nBase = nh * 128;
  const int g = lane >> 4, nlo = lane & 15;

  // ---- scores: S[32,256] = Q @ K^T * rscale + tb ----
  float s[8][8];
  {
    const v16h a = load_a_frag(Qsh, 40, mt * 16, 0, lane);  // K(=Dh)=32: one frag
#pragma unroll
    for (int nt = 0; nt < 8; ++nt) {
      const v16h bf = load_b_frag_nmajor(Ksh, 40, nBase + nt * 16, 0, lane);
      v8f c = {};
      c = wmma_f16(a, bf, c);
      const float bias = tbs[nBase + nt * 16 + nlo];
#pragma unroll
      for (int v = 0; v < 8; ++v) s[nt][v] = c[v] * RSCALE + bias;
    }
  }

  // ---- softmax over k (256): shuffle reduce in 16-lane group, LDS across waves
#pragma unroll
  for (int v = 0; v < 8; ++v) {
    float m = s[0][v];
#pragma unroll
    for (int nt = 1; nt < 8; ++nt) m = fmaxf(m, s[nt][v]);
#pragma unroll
    for (int msk = 8; msk >= 1; msk >>= 1) m = fmaxf(m, __shfl_xor(m, msk, 32));
    if (nlo == 0) redm[mt * 16 + v + 8 * g][nh] = m;
  }
  __syncthreads();
#pragma unroll
  for (int v = 0; v < 8; ++v) {
    const int row = mt * 16 + v + 8 * g;
    const float m = fmaxf(redm[row][0], redm[row][1]);
    float sm = 0.f;
#pragma unroll
    for (int nt = 0; nt < 8; ++nt) {
      s[nt][v] = __expf(s[nt][v] - m);
      sm += s[nt][v];
    }
#pragma unroll
    for (int msk = 8; msk >= 1; msk >>= 1) sm += __shfl_xor(sm, msk, 32);
    if (nlo == 0) reds[row][nh] = sm;
  }
  __syncthreads();
#pragma unroll
  for (int v = 0; v < 8; ++v) {
    const int row = mt * 16 + v + 8 * g;
    const float inv = 1.0f / (reds[row][0] + reds[row][1]);
#pragma unroll
    for (int nt = 0; nt < 8; ++nt)
      Psh[row * 264 + nBase + nt * 16 + nlo] = (_Float16)(s[nt][v] * inv);
  }
  __syncthreads();

  // ---- O[32,32] = P[32,256] @ V[256,32]; wave -> one 16x16 output tile ----
  const int ntv = wid & 1;
  v8f acc = {};
#pragma unroll
  for (int k0 = 0; k0 < TA_S; k0 += 32) {
    const v16h ap = load_a_frag(Psh, 264, mt * 16, k0, lane);
    const v16h bv = load_b_frag_kmajor(Vsh, 40, k0, ntv * 16, lane);
    acc = wmma_f16(ap, bv, acc);
  }
  // sigmoid gate fused into the store
#pragma unroll
  for (int v = 0; v < 8; ++v) {
    const int jloc = mt * 16 + v + 8 * g;
    const int d = ntv * 16 + nlo;
    const size_t gi = (rowbase + j0 + jloc) * TA_D + h * TA_DH + d;
    const float gv = (float)Gb[gi];
    const float sig = 1.0f / (1.0f + __expf(-gv));
    OG[gi] = (_Float16)(acc[v] * sig);
  }
}

// ---------------------------------------------------------------------------
// Host-side launcher
// ---------------------------------------------------------------------------
extern "C" void kernel_launch(void* const* d_in, const int* in_sizes, int n_in,
                              void* d_out, int out_size, void* d_ws,
                              size_t ws_size, hipStream_t stream) {
  (void)in_sizes; (void)n_in; (void)out_size; (void)ws_size;
  const float* x      = (const float*)d_in[0];
  const float* w_ln   = (const float*)d_in[1];
  const float* b_ln   = (const float*)d_in[2];
  const float* w_bias = (const float*)d_in[3];
  const float* w_q    = (const float*)d_in[4];
  const float* w_k    = (const float*)d_in[5];
  const float* w_v    = (const float*)d_in[6];
  const float* w_g    = (const float*)d_in[7];
  const float* w_o    = (const float*)d_in[8];
  float* out = (float*)d_out;

  const size_t R = (size_t)TA_S * TA_S;  // 65536 rows
  char* ws = (char*)d_ws;
  size_t off = 0;
  auto take = [&](size_t bytes) -> char* {
    char* p = ws + off;
    off += (bytes + 255) & ~(size_t)255;
    return p;
  };
  _Float16* Xn  = (_Float16*)take(R * TA_D * 2);
  _Float16* Yq  = (_Float16*)take(R * TA_D * 2);
  _Float16* Yk  = (_Float16*)take(R * TA_D * 2);
  _Float16* Yv  = (_Float16*)take(R * TA_D * 2);
  _Float16* Yg  = (_Float16*)take(R * TA_D * 2);
  _Float16* OGb = (_Float16*)take(R * TA_D * 2);
  float*    TB  = (float*)take((size_t)TA_H * R * 4);
  _Float16* wq16 = (_Float16*)take(16384 * 2);
  _Float16* wk16 = (_Float16*)take(16384 * 2);
  _Float16* wv16 = (_Float16*)take(16384 * 2);
  _Float16* wg16 = (_Float16*)take(16384 * 2);
  _Float16* wo16 = (_Float16*)take(16384 * 2);
  _Float16* wb16 = (_Float16*)take(512 * 2);

  cvt_kernel<<<64, 256, 0, stream>>>(w_q, w_k, w_v, w_g, w_o, w_bias, wq16,
                                     wk16, wv16, wg16, wo16, wb16);
  ln_kernel<<<R / 8, 256, 0, stream>>>(x, w_ln, b_ln, Xn);

  dim3 gg(R / 128, TA_D / 64);  // (512, 2)
  gemm128<_Float16><<<gg, 256, 0, stream>>>(Xn, wq16, Yq, TA_D);
  gemm128<_Float16><<<gg, 256, 0, stream>>>(Xn, wk16, Yk, TA_D);
  gemm128<_Float16><<<gg, 256, 0, stream>>>(Xn, wv16, Yv, TA_D);
  gemm128<_Float16><<<gg, 256, 0, stream>>>(Xn, wg16, Yg, TA_D);
  tb_kernel<<<(TA_H * R) / 256, 256, 0, stream>>>(Xn, wb16, TB);

  attn_kernel<<<TA_S * TA_H * 8, 128, 0, stream>>>(Yq, Yk, Yv, Yg, TB, OGb);

  gemm128<float><<<gg, 256, 0, stream>>>(OGb, wo16, out, TA_D);
}
